// EncoderRNN_85014582657145
// MI455X (gfx1250) — matrix operand order repaired and checked by reading, not compiled, not run on hardware
//
#include <hip/hip_runtime.h>
#include <hip/hip_bf16.h>
#include <math.h>

// Problem constants
#define VSZ 50000
#define E   256
#define H   256
#define DHD 512
#define NC  5
#define BB  64
#define TT  2048
#define G   768   // 3*H gates

typedef __attribute__((ext_vector_type(16))) __bf16 v16bf;
typedef __attribute__((ext_vector_type(8)))  float  v8f;

union V16 { v16bf v; unsigned int u[8]; unsigned short s[16]; };

__device__ __forceinline__ unsigned short f2bf(float f) {
    unsigned int u = __float_as_uint(f);
    u += 0x7FFFu + ((u >> 16) & 1u);   // round-to-nearest-even
    return (unsigned short)(u >> 16);
}
__device__ __forceinline__ float bf2f(unsigned short s) {
    return __uint_as_float(((unsigned int)s) << 16);
}

// B-fragment (32x16, bf16) loader from a row-major [N][K=256] bf16 weight
// matrix. Lane L: n = L&15, K = (L>>4)*8 + e (e<8), 16 + (L>>4)*8 + (e-8) (e>=8).
__device__ __forceinline__ v16bf load_b_g(const unsigned short* __restrict__ W,
                                          int col, int kbase) {
    const uint4* p = reinterpret_cast<const uint4*>(W + (size_t)col * H + kbase);
    uint4 q0 = p[0], q1 = p[2];
    V16 b;
    b.u[0] = q0.x; b.u[1] = q0.y; b.u[2] = q0.z; b.u[3] = q0.w;
    b.u[4] = q1.x; b.u[5] = q1.y; b.u[6] = q1.z; b.u[7] = q1.w;
    return b.v;
}

// ---------------------------------------------------------------------------
// Kernel 1: convert the four GRU weight matrices f32 -> bf16 into workspace.
// Layout in dst: [W_ih_f][W_hh_f][W_ih_b][W_hh_b], each 768*256.
// ---------------------------------------------------------------------------
__global__ void __launch_bounds__(256, 1)
k_convert(const float* __restrict__ a, const float* __restrict__ b,
          const float* __restrict__ c, const float* __restrict__ d,
          unsigned short* __restrict__ o) {
    const int N1 = G * H;
    int i = blockIdx.x * 256 + threadIdx.x;      // i < 4*N1
    int m = i / N1, r = i - m * N1;
    float v;
    if      (m == 0) v = a[r];
    else if (m == 1) v = b[r];
    else if (m == 2) v = c[r];
    else             v = d[r];
    o[i] = f2bf(v);
}

// ---------------------------------------------------------------------------
// Kernel 2: fused embedding-gather + mask + input-projection GEMM.
// gi[dir][t][b][g] (bf16) = (mask * emb[seqs[b][t]]) @ W_ih[dir]^T
// One block per (timestep t, direction). Block = 256 thr = 8 waves.
// Each wave owns 96 gate columns; processed in two 3-n-tile passes to keep
// only 96 accumulator VGPRs live. B fragments loaded once each.
// __launch_bounds__(256,1): 2 waves/SIMD -> large per-wave VGPR budget.
// ---------------------------------------------------------------------------
__global__ void __launch_bounds__(256, 1)
k_gi_gemm(const int* __restrict__ seqs, const int* __restrict__ lens,
          const float* __restrict__ emb,
          const unsigned short* __restrict__ wih_f,
          const unsigned short* __restrict__ wih_b,
          unsigned short* __restrict__ gi_f,
          unsigned short* __restrict__ gi_b) {
    __shared__ unsigned short xs[64][272];       // 64 rows x 256 K, padded

    const int t   = blockIdx.x;
    const int dir = blockIdx.y;
    const unsigned short* W  = dir ? wih_b : wih_f;
    unsigned short*       gi = dir ? gi_b  : gi_f;
    const int tid = threadIdx.x;

    // Stage A: gather+mask+convert embedding rows for timestep t, all batches.
    {
        int b  = tid >> 2;
        int k0 = (tid & 3) * 64;
        int tok   = seqs[b * TT + t];
        int valid = (t >= TT - lens[b]) ? 1 : 0;
        const float* er = emb + (size_t)tok * E + k0;
        #pragma unroll
        for (int k = 0; k < 64; k += 4) {
            float4 v = *reinterpret_cast<const float4*>(er + k);
            if (!valid) { v.x = 0.f; v.y = 0.f; v.z = 0.f; v.w = 0.f; }
            xs[b][k0 + k + 0] = f2bf(v.x);
            xs[b][k0 + k + 1] = f2bf(v.y);
            xs[b][k0 + k + 2] = f2bf(v.z);
            xs[b][k0 + k + 3] = f2bf(v.w);
        }
    }
    __syncthreads();

    const int wave = tid >> 5, lane = tid & 31;
    const int ln = lane & 15, hi = lane >> 4;
    const int colbase = wave * 96;

    #pragma unroll 1
    for (int half = 0; half < 2; ++half) {
        v8f acc[4][3];
        #pragma unroll
        for (int mt = 0; mt < 4; ++mt)
            #pragma unroll
            for (int nt = 0; nt < 3; ++nt)
                acc[mt][nt] = (v8f){0.f,0.f,0.f,0.f,0.f,0.f,0.f,0.f};

        #pragma unroll
        for (int ks = 0; ks < 8; ++ks) {
            // A fragments (16x32 bf16) from LDS, ISA-interleaved K layout.
            V16 afrag[4];
            #pragma unroll
            for (int mt = 0; mt < 4; ++mt) {
                int row = mt * 16 + ln;
                const uint4* p0 = reinterpret_cast<const uint4*>(&xs[row][ks * 32 + hi * 8]);
                const uint4* p1 = reinterpret_cast<const uint4*>(&xs[row][ks * 32 + 16 + hi * 8]);
                uint4 q0 = *p0, q1 = *p1;
                afrag[mt].u[0] = q0.x; afrag[mt].u[1] = q0.y;
                afrag[mt].u[2] = q0.z; afrag[mt].u[3] = q0.w;
                afrag[mt].u[4] = q1.x; afrag[mt].u[5] = q1.y;
                afrag[mt].u[6] = q1.z; afrag[mt].u[7] = q1.w;
            }
            #pragma unroll
            for (int nt = 0; nt < 3; ++nt) {
                v16bf bfrag = load_b_g(W, colbase + (half * 3 + nt) * 16 + ln,
                                       ks * 32 + hi * 8);
                #pragma unroll
                for (int mt = 0; mt < 4; ++mt)
                    acc[mt][nt] = __builtin_amdgcn_wmma_f32_16x16x32_bf16(
                        false, afrag[mt].v, false, bfrag, (short)0, acc[mt][nt],
                        false, false);
            }
        }

        // Epilogue: D layout n = ln, m = mt*16 + 8*hi + r. Store bf16.
        #pragma unroll
        for (int mt = 0; mt < 4; ++mt)
            #pragma unroll
            for (int nt = 0; nt < 3; ++nt) {
                int col = colbase + (half * 3 + nt) * 16 + ln;
                #pragma unroll
                for (int r = 0; r < 8; ++r) {
                    int m = mt * 16 + hi * 8 + r;
                    gi[((size_t)t * BB + m) * G + col] = f2bf(acc[mt][nt][r]);
                }
            }
    }
}

// ---------------------------------------------------------------------------
// Kernel 3: persistent GRU scan. grid.x = 2 (fwd/bwd), block = 256 = 8 waves.
// Wave w owns hidden slice j in [w*32, w*32+32). Each step is two jt-passes
// (16 hidden cols each, all 3 gates -> acc[4][3] = 96 VGPRs). The bf16 hidden
// state is double-buffered in LDS: pass reads hbf[p], the in-register GRU cell
// update writes hbf[1-p]; one s_barrier per step. W_hh streamed bf16 from L2
// (each fragment exactly once per step).
// ---------------------------------------------------------------------------
__global__ void __launch_bounds__(256, 1)
k_scan(const unsigned short* __restrict__ gi_f,
       const unsigned short* __restrict__ gi_b,
       const unsigned short* __restrict__ whh_f,
       const unsigned short* __restrict__ whh_b,
       const float* __restrict__ b_ih_f, const float* __restrict__ b_hh_f,
       const float* __restrict__ b_ih_b, const float* __restrict__ b_hh_b,
       float* __restrict__ rep) {
    __shared__ unsigned short hbf[2][64][272];   // double-buffered bf16 h

    const int dir = blockIdx.x;
    const unsigned short* gi  = dir ? gi_b  : gi_f;
    const unsigned short* Whh = dir ? whh_b : whh_f;
    const float* bih = dir ? b_ih_b : b_ih_f;
    const float* bhh = dir ? b_hh_b : b_hh_f;

    const int tid = threadIdx.x, wave = tid >> 5, lane = tid & 31;
    const int ln = lane & 15, hi = lane >> 4;

    for (int i = tid; i < 2 * 64 * 272; i += 256)
        (&hbf[0][0][0])[i] = 0;

    // Per-lane bias registers: j = wave*32 + jt*16 + ln, gates r/z/n.
    float bi[3][2], bh[3][2];
    #pragma unroll
    for (int g = 0; g < 3; ++g)
        #pragma unroll
        for (int jt = 0; jt < 2; ++jt) {
            int j = wave * 32 + jt * 16 + ln;
            bi[g][jt] = bih[g * 256 + j];
            bh[g][jt] = bhh[g * 256 + j];
        }

    float h[4][2][8];
    #pragma unroll
    for (int mt = 0; mt < 4; ++mt)
        #pragma unroll
        for (int jt = 0; jt < 2; ++jt)
            #pragma unroll
            for (int r = 0; r < 8; ++r) h[mt][jt][r] = 0.f;

    __syncthreads();

    #pragma unroll 1
    for (int step = 0; step < TT; ++step) {
        const int tg = dir ? (TT - 1 - step) : step;
        const int p  = step & 1;
        const unsigned short* gslab = gi + (size_t)tg * BB * G;

        #pragma unroll 1
        for (int jt = 0; jt < 2; ++jt) {
            // ---- Pass: gh = h @ W_hh^T for 3 gates x 16 hidden columns ----
            v8f acc[4][3];
            #pragma unroll
            for (int mt = 0; mt < 4; ++mt)
                #pragma unroll
                for (int g = 0; g < 3; ++g)
                    acc[mt][g] = (v8f){0.f,0.f,0.f,0.f,0.f,0.f,0.f,0.f};

            #pragma unroll
            for (int ks = 0; ks < 8; ++ks) {
                V16 afrag[4];
                #pragma unroll
                for (int mt = 0; mt < 4; ++mt) {
                    int row = mt * 16 + ln;
                    const uint4* p0 = reinterpret_cast<const uint4*>(
                        &hbf[p][row][ks * 32 + hi * 8]);
                    const uint4* p1 = reinterpret_cast<const uint4*>(
                        &hbf[p][row][ks * 32 + 16 + hi * 8]);
                    uint4 q0 = *p0, q1 = *p1;
                    afrag[mt].u[0] = q0.x; afrag[mt].u[1] = q0.y;
                    afrag[mt].u[2] = q0.z; afrag[mt].u[3] = q0.w;
                    afrag[mt].u[4] = q1.x; afrag[mt].u[5] = q1.y;
                    afrag[mt].u[6] = q1.z; afrag[mt].u[7] = q1.w;
                }
                #pragma unroll
                for (int g = 0; g < 3; ++g) {
                    int col = g * 256 + wave * 32 + jt * 16 + ln;
                    v16bf bfrag = load_b_g(Whh, col, ks * 32 + hi * 8);
                    #pragma unroll
                    for (int mt = 0; mt < 4; ++mt)
                        acc[mt][g] = __builtin_amdgcn_wmma_f32_16x16x32_bf16(
                            false, afrag[mt].v, false, bfrag, (short)0,
                            acc[mt][g], false, false);
                }
            }

            // ---- GRU cell update for this jt slice (registers only) ----
            const int j = wave * 32 + jt * 16 + ln;
            #pragma unroll
            for (int mt = 0; mt < 4; ++mt)
                #pragma unroll
                for (int r = 0; r < 8; ++r) {
                    int brow = mt * 16 + hi * 8 + r;
                    const unsigned short* grow = gslab + (size_t)brow * G;
                    float gir = bf2f(grow[j])       + bi[0][jt];
                    float giz = bf2f(grow[256 + j]) + bi[1][jt];
                    float gin = bf2f(grow[512 + j]) + bi[2][jt];
                    float ghr = acc[mt][0][r] + bh[0][jt];
                    float ghz = acc[mt][1][r] + bh[1][jt];
                    float ghn = acc[mt][2][r] + bh[2][jt];
                    float rg = 1.f / (1.f + __expf(-(gir + ghr)));
                    float zg = 1.f / (1.f + __expf(-(giz + ghz)));
                    float ng = tanhf(gin + rg * ghn);
                    float hv = (1.f - zg) * ng + zg * h[mt][jt][r];
                    h[mt][jt][r] = hv;
                    hbf[1 - p][brow][j] = f2bf(hv);
                }
        }

        // Prefetch next step's gi slab (96KB) while waiting at the barrier.
        if (step + 1 < TT) {
            int tn = dir ? (TT - 2 - step) : (step + 1);
            const char* pc = (const char*)(gi + (size_t)tn * BB * G) + tid * 384;
            __builtin_prefetch(pc +   0, 0, 1);
            __builtin_prefetch(pc + 128, 0, 1);
            __builtin_prefetch(pc + 256, 0, 1);
        }

        __syncthreads();   // hbf[1-p] fully written before next step reads it
    }

    // Final hidden -> rep[b][dir*256 + j]
    #pragma unroll
    for (int mt = 0; mt < 4; ++mt)
        #pragma unroll
        for (int jt = 0; jt < 2; ++jt)
            #pragma unroll
            for (int r = 0; r < 8; ++r) {
                int brow = mt * 16 + hi * 8 + r;
                int j = wave * 32 + jt * 16 + ln;
                rep[(size_t)brow * (2 * H) + dir * H + j] = h[mt][jt][r];
            }
}

// ---------------------------------------------------------------------------
// Kernel 4/5: MLP head (tiny; f32 VALU).
// ---------------------------------------------------------------------------
__global__ void __launch_bounds__(256, 1)
k_head1(const float* __restrict__ rep, const float* __restrict__ W1,
        const float* __restrict__ b1, float* __restrict__ h1) {
    int idx = blockIdx.x * 256 + threadIdx.x;    // 64*512
    int b = idx >> 9, d = idx & 511;
    const float* rr = rep + (size_t)b * 512;
    const float* wr = W1 + (size_t)d * 512;
    float s = b1[d];
    for (int k = 0; k < 512; k += 4) {
        float4 x = *reinterpret_cast<const float4*>(rr + k);
        float4 w = *reinterpret_cast<const float4*>(wr + k);
        s += x.x * w.x + x.y * w.y + x.z * w.z + x.w * w.w;
    }
    h1[idx] = fmaxf(s, 0.f);
}

__global__ void __launch_bounds__(64, 1)
k_head2(const float* __restrict__ h1, const float* __restrict__ W2,
        const float* __restrict__ b2, float* __restrict__ out) {
    int b = threadIdx.x;
    if (b >= BB) return;
    float lg[NC];
    #pragma unroll
    for (int c = 0; c < NC; ++c) {
        const float* hr = h1 + (size_t)b * DHD;
        const float* wr = W2 + (size_t)c * DHD;
        float s = b2[c];
        for (int k = 0; k < DHD; k += 4) {
            float4 x = *reinterpret_cast<const float4*>(hr + k);
            float4 w = *reinterpret_cast<const float4*>(wr + k);
            s += x.x * w.x + x.y * w.y + x.z * w.z + x.w * w.w;
        }
        lg[c] = s;
    }
    float mx = lg[0];
    #pragma unroll
    for (int c = 1; c < NC; ++c) mx = fmaxf(mx, lg[c]);
    float ex[NC], den = 0.f;
    #pragma unroll
    for (int c = 0; c < NC; ++c) { ex[c] = __expf(lg[c] - mx); den += ex[c]; }
    float inv = 1.f / den;
    #pragma unroll
    for (int c = 0; c < NC; ++c) out[b * NC + c] = ex[c] * inv;
}

// ---------------------------------------------------------------------------
extern "C" void kernel_launch(void* const* d_in, const int* in_sizes, int n_in,
                              void* d_out, int out_size, void* d_ws, size_t ws_size,
                              hipStream_t stream) {
    (void)in_sizes; (void)n_in; (void)out_size; (void)ws_size;

    const int*   seqs   = (const int*)d_in[0];
    const int*   lens   = (const int*)d_in[1];
    const float* emb    = (const float*)d_in[2];
    const float* W_ih_f = (const float*)d_in[3];
    const float* W_hh_f = (const float*)d_in[4];
    const float* b_ih_f = (const float*)d_in[5];
    const float* b_hh_f = (const float*)d_in[6];
    const float* W_ih_b = (const float*)d_in[7];
    const float* W_hh_b = (const float*)d_in[8];
    const float* b_ih_b = (const float*)d_in[9];
    const float* b_hh_b = (const float*)d_in[10];
    const float* W1     = (const float*)d_in[11];
    const float* b1     = (const float*)d_in[12];
    const float* W2     = (const float*)d_in[13];
    const float* b2     = (const float*)d_in[14];
    float* out = (float*)d_out;

    char* ws = (char*)d_ws;
    size_t o = 0;
    auto take = [&](size_t bytes) -> char* {
        char* p = ws + o;
        o += (bytes + 255) & ~(size_t)255;
        return p;
    };

    const size_t WN = (size_t)G * H;                       // 768*256
    unsigned short* wbf = (unsigned short*)take(4 * WN * 2);
    unsigned short* gif = (unsigned short*)take((size_t)TT * BB * G * 2);  // ~192MB
    unsigned short* gib = (unsigned short*)take((size_t)TT * BB * G * 2);  // ~192MB
    float* rep = (float*)take((size_t)BB * 2 * H * 4);
    float* h1  = (float*)take((size_t)BB * DHD * 4);

    unsigned short* wihf = wbf;
    unsigned short* whhf = wbf + WN;
    unsigned short* wihb = wbf + 2 * WN;
    unsigned short* whhb = wbf + 3 * WN;

    k_convert<<<dim3((unsigned)(4 * WN / 256)), dim3(256), 0, stream>>>(
        W_ih_f, W_hh_f, W_ih_b, W_hh_b, wbf);

    k_gi_gemm<<<dim3(TT, 2), dim3(256), 0, stream>>>(
        seqs, lens, emb, wihf, wihb, gif, gib);

    k_scan<<<dim3(2), dim3(256), 0, stream>>>(
        gif, gib, whhf, whhb, b_ih_f, b_hh_f, b_ih_b, b_hh_b, rep);

    k_head1<<<dim3(BB * DHD / 256), dim3(256), 0, stream>>>(rep, W1, b1, h1);
    k_head2<<<dim3(1), dim3(64), 0, stream>>>(h1, W2, b2, out);
}
